// VideoCrossAttentionWith3DRope_51866025067095
// MI455X (gfx1250) — matrix operand order repaired and checked by reading, not compiled, not run on hardware
//
#include <hip/hip_runtime.h>

typedef __bf16 bf16;
typedef __attribute__((ext_vector_type(16))) __bf16 bf16x16;
typedef __attribute__((ext_vector_type(8)))  __bf16 bf16x8;
typedef __attribute__((ext_vector_type(4)))  __bf16 bf16x4;
typedef __attribute__((ext_vector_type(8)))  float  f32x8;
typedef __attribute__((ext_vector_type(4)))  float  f32x4;
typedef __attribute__((ext_vector_type(4)))  int    i32x4;
typedef __attribute__((address_space(1))) i32x4 as1_i32x4;
typedef __attribute__((address_space(3))) i32x4 as3_i32x4;

#define NHEAD   8
#define HDIM    128
#define DMODEL  1024
#define NQTOK   2048
#define NKTOK   32768
#define TQF     8
#define SQL     256
#define RSK     4096          // keys per query frame (r*SK = 4*1024)
#define NKB     (RSK / 32)    // 128 key tiles
#define SM_SCALE 0.08838834764831845f   // 1/sqrt(128)
#define LOG2_10000 13.287712379549449f

// Build a 16-element bf16 WMMA fragment from two contiguous 8-element (16B) loads.
__device__ __forceinline__ bf16x16 ld_frag16(const bf16* p0, const bf16* p1) {
  bf16x8 lo = *(const bf16x8*)p0;
  bf16x8 hi = *(const bf16x8*)p1;
  bf16x16 r;
#pragma unroll
  for (int i = 0; i < 8; ++i) { r[i] = lo[i]; r[i + 8] = hi[i]; }
  return r;
}

__device__ __forceinline__ f32x8 wmma_bf16(bf16x16 a, bf16x16 b, f32x8 c) {
  return __builtin_amdgcn_wmma_f32_16x16x32_bf16(false, a, false, b, (short)0, c,
                                                 false, false);
}

// gfx1250 async global->LDS copy (ASYNCcnt path), with sync fallback.
__device__ __forceinline__ void cp16_async(const bf16* g, bf16* l) {
#if __has_builtin(__builtin_amdgcn_global_load_async_to_lds_b128)
  __builtin_amdgcn_global_load_async_to_lds_b128(
      (as1_i32x4*)const_cast<bf16*>(g), (as3_i32x4*)l, 0, 0);
#else
  *(bf16x8*)l = *(const bf16x8*)g;
#endif
}

__device__ __forceinline__ void wait_async_all() {
#if __has_builtin(__builtin_amdgcn_s_wait_asynccnt)
  __builtin_amdgcn_s_wait_asynccnt(0);
#elif __has_builtin(__builtin_amdgcn_global_load_async_to_lds_b128)
  asm volatile("s_wait_asynccnt 0" ::: "memory");
#endif
}

// ---------------------------------------------------------------------------
// W transpose + convert: WT[c][k] (bf16) = W[k][c].  Read-coalesced.
// ---------------------------------------------------------------------------
__global__ __launch_bounds__(256) void wtr_kernel(const float* __restrict__ W,
                                                  bf16* __restrict__ WT) {
  int idx = blockIdx.x * 256 + threadIdx.x;   // over 1024*1024
  int k = idx >> 10;
  int c = idx & 1023;
  WT[((size_t)c << 10) + k] = (bf16)W[idx];
}

// ---------------------------------------------------------------------------
// Phase 1: Y = (X @ W + b) with fused 3D mRoPE for Q/K.
// Block: 128 thr (4 waves), 64 tokens. Whole 64x1024 X slice staged ONCE in
// LDS as bf16 (read-once HBM traffic); loop over all 8 heads internally.
// which: 0=Q (slow pos, row-major out), 1=K (fast pos, row-major out),
//        2=V (no rope, TRANSPOSED out [head][d][tok])
// ---------------------------------------------------------------------------
__global__ __launch_bounds__(128) void qkv_rope_kernel(
    const float* __restrict__ X, const bf16* __restrict__ WT,
    const float* __restrict__ bias, bf16* __restrict__ Y,
    int ntok, int which)
{
  __shared__ __align__(16) bf16 Xs[64][1032];   // padded: conflict-free frags
  __shared__ __align__(16) bf16 Bs[128][40];    // [col][k], padded

  const int tid  = threadIdx.x;
  const int wave = tid >> 5;
  const int lane = tid & 31;
  const int lh   = lane >> 4;
  const int lc   = lane & 15;
  const int row0 = blockIdx.x * 64;

  // stage full X slice (64 x 1024), fp32 -> bf16
  for (int i = 0; i < 128; ++i) {
    int id = tid + i * 128;          // 0..16383 float4 chunks
    int r  = id >> 8;                // token row 0..63
    int c4 = id & 255;               // float4 group
    f32x4 x4 = *(const f32x4*)(X + (size_t)(row0 + r) * DMODEL + c4 * 4);
    bf16x4 b4;
#pragma unroll
    for (int u = 0; u < 4; ++u) b4[u] = (bf16)x4[u];
    *(bf16x4*)&Xs[r][c4 * 4] = b4;
  }
  __syncthreads();

  for (int head = 0; head < NHEAD; ++head) {
    f32x8 acc[8];
#pragma unroll
    for (int nt = 0; nt < 8; ++nt)
#pragma unroll
      for (int j = 0; j < 8; ++j) acc[nt][j] = 0.f;

    for (int kk = 0; kk < DMODEL; kk += 32) {
      // stage W^T tile (128 cols x 32 k), already bf16: pure 16B copies
#pragma unroll
      for (int i = 0; i < 4; ++i) {
        int id = tid + i * 128;      // 0..511
        int cc = id >> 2;            // col 0..127
        int k8 = id & 3;             // 8-elt k group
        *(bf16x8*)&Bs[cc][k8 * 8] =
            *(const bf16x8*)(WT + ((size_t)(head * 128 + cc) << 10) + kk + k8 * 8);
      }
      __syncthreads();

      const bf16* arow = &Xs[wave * 16 + lc][kk];
      bf16x16 a = ld_frag16(arow + lh * 8, arow + 16 + lh * 8);
#pragma unroll
      for (int nt = 0; nt < 8; ++nt) {
        const bf16* brow = &Bs[nt * 16 + lc][0];
        acc[nt] = wmma_bf16(a, ld_frag16(brow + lh * 8, brow + 16 + lh * 8), acc[nt]);
      }
      __syncthreads();
    }

    const float* bv = bias + head * 128;

    if (which == 2) {   // V: bias + transposed store [head][d][tok]
#pragma unroll
      for (int nt = 0; nt < 8; ++nt) {
        int c = nt * 16 + lc;
        float bb = bv[c];
#pragma unroll
        for (int j = 0; j < 8; ++j) {
          int tok = row0 + wave * 16 + j + 8 * lh;
          Y[((size_t)(head * HDIM + c)) * ntok + tok] = (bf16)(acc[nt][j] + bb);
        }
      }
      continue;
    }

    // Q/K: fused 3D mRoPE. Channel c pairs with c+64 (same lane, tile nt^4).
#pragma unroll
    for (int nt = 0; nt < 4; ++nt) {
      int c0 = nt * 16 + lc;          // 0..63
      int c1 = c0 + 64;
      float b0 = bv[c0], b1 = bv[c1];
      int axis = (c0 < 16) ? 0 : (c0 < 40) ? 1 : 2;
      float inv = exp2f((float)c0 * (-LOG2_10000 / 64.f));  // 10000^(-c0/64)
#pragma unroll
      for (int j = 0; j < 8; ++j) {
        int tok = row0 + wave * 16 + j + 8 * lh;
        int pt, ph, pw;
        if (which == 0) {             // slow (query) positions
          int t = tok >> 8, s = tok & 255;
          pt = (t * 31) / 7;          // == linspace(0,31,8).astype(int)
          ph = 2 * (s >> 4) + 1;
          pw = 2 * (s & 15) + 1;
        } else {                      // fast (key) positions: identity grid
          pt = tok >> 10;
          int s = tok & 1023;
          ph = s >> 5;
          pw = s & 31;
        }
        int p = (axis == 0) ? pt : (axis == 1) ? ph : pw;
        float ang = (float)p * inv;
        float cs = __cosf(ang), sn = __sinf(ang);
        float x0 = acc[nt][j] + b0;
        float x1 = acc[nt + 4][j] + b1;
        float y0 = x0 * cs - x1 * sn;
        float y1 = x1 * cs + x0 * sn;
        size_t base = ((size_t)head * ntok + tok) * HDIM;
        Y[base + c0] = (bf16)y0;
        Y[base + c1] = (bf16)y1;
      }
    }
  }
}

// ---------------------------------------------------------------------------
// Phase 2: grouped flash attention, double-buffered async LDS staging.
// Block = 128 thr (4 waves), 64 queries. grid = (4, NHEAD, TQF).
// ---------------------------------------------------------------------------
__global__ __launch_bounds__(128) void attn_kernel(
    const bf16* __restrict__ Qw, const bf16* __restrict__ Kw,
    const bf16* __restrict__ Vw, float* __restrict__ out)
{
  __shared__ __align__(16) bf16 Ks[2][32][136];   // [key][d], padded
  __shared__ __align__(16) bf16 Vt[2][128][40];   // [d][key], padded
  __shared__ __align__(16) bf16 Ps[4][16][40];    // per-wave P tile, padded

  const int tid  = threadIdx.x;
  const int wave = tid >> 5;
  const int lane = tid & 31;
  const int lh   = lane >> 4;
  const int lc   = lane & 15;
  const int qblk = blockIdx.x;
  const int head = blockIdx.y;
  const int tq   = blockIdx.z;

  const size_t kbase = ((size_t)head * NKTOK + (size_t)tq * RSK) * HDIM;        // K row-major
  const size_t vbase = ((size_t)head * HDIM + tid) * NKTOK + (size_t)tq * RSK;  // V [d][tok], d=tid

  // Q fragments for this wave's 16 rows (4 k-steps over d=128), loaded once.
  const bf16* qbase =
      Qw + ((size_t)head * NQTOK + tq * SQL + qblk * 64 + wave * 16 + lc) * HDIM;
  bf16x16 qf[4];
#pragma unroll
  for (int ks = 0; ks < 4; ++ks)
    qf[ks] = ld_frag16(qbase + ks * 32 + lh * 8, qbase + ks * 32 + 16 + lh * 8);

  f32x8 oacc[8];
#pragma unroll
  for (int nt = 0; nt < 8; ++nt)
#pragma unroll
    for (int j = 0; j < 8; ++j) oacc[nt][j] = 0.f;
  float mrow[8], lrow[8];
#pragma unroll
  for (int j = 0; j < 8; ++j) { mrow[j] = -1e30f; lrow[j] = 0.f; }

  auto stage = [&](int kb, int buf) {
    // K tile: 32 keys x 128 d, coalesced 16B chunks
#pragma unroll
    for (int i = 0; i < 4; ++i) {
      int id = tid + i * 128;
      int r  = id >> 4;
      int c8 = id & 15;
      cp16_async(Kw + kbase + (size_t)(kb * 32 + r) * HDIM + c8 * 8,
                 &Ks[buf][r][c8 * 8]);
    }
    // V tile: thread t copies its d-row (32 keys = 64B) from transposed Vw
#pragma unroll
    for (int i = 0; i < 4; ++i)
      cp16_async(Vw + vbase + kb * 32 + i * 8, &Vt[buf][tid][i * 8]);
  };

  stage(0, 0);
  wait_async_all();
  __syncthreads();

  for (int kb = 0; kb < NKB; ++kb) {
    const int buf = kb & 1;
    if (kb + 1 < NKB) stage(kb + 1, buf ^ 1);   // prefetch overlaps compute

    // S = Q K^T (16 rows x 32 keys), f32 accum
    f32x8 s[2];
#pragma unroll
    for (int nt = 0; nt < 2; ++nt)
#pragma unroll
      for (int j = 0; j < 8; ++j) s[nt][j] = 0.f;
#pragma unroll
    for (int ks = 0; ks < 4; ++ks)
#pragma unroll
      for (int nt = 0; nt < 2; ++nt) {
        const bf16* kr = &Ks[buf][nt * 16 + lc][ks * 32 + lh * 8];
        s[nt] = wmma_bf16(qf[ks], ld_frag16(kr, kr + 16), s[nt]);
      }

    // online softmax: rows m = j + 8*lh; reduce across 16-lane halves
#pragma unroll
    for (int j = 0; j < 8; ++j) {
      float s0 = s[0][j] * SM_SCALE, s1 = s[1][j] * SM_SCALE;
      float mt = fmaxf(s0, s1);
      mt = fmaxf(mt, __shfl_xor(mt, 1, 32));
      mt = fmaxf(mt, __shfl_xor(mt, 2, 32));
      mt = fmaxf(mt, __shfl_xor(mt, 4, 32));
      mt = fmaxf(mt, __shfl_xor(mt, 8, 32));
      float mnew  = fmaxf(mrow[j], mt);
      float alpha = __expf(mrow[j] - mnew);
      float p0 = __expf(s0 - mnew), p1 = __expf(s1 - mnew);
      float ps = p0 + p1;
      ps += __shfl_xor(ps, 1, 32);
      ps += __shfl_xor(ps, 2, 32);
      ps += __shfl_xor(ps, 4, 32);
      ps += __shfl_xor(ps, 8, 32);
      lrow[j] = lrow[j] * alpha + ps;
      mrow[j] = mnew;
#pragma unroll
      for (int nt = 0; nt < 8; ++nt) oacc[nt][j] *= alpha;
      int m = j + 8 * lh;
      Ps[wave][m][lc]      = (bf16)p0;
      Ps[wave][m][16 + lc] = (bf16)p1;
    }

    // O += P V  (P via per-wave LDS round-trip; DS ops in-order per wave)
    const bf16* pr = &Ps[wave][lc][0];
    bf16x16 pa = ld_frag16(pr + lh * 8, pr + 16 + lh * 8);
#pragma unroll
    for (int nt = 0; nt < 8; ++nt) {
      const bf16* vr = &Vt[buf][nt * 16 + lc][lh * 8];
      oacc[nt] = wmma_bf16(pa, ld_frag16(vr, vr + 16), oacc[nt]);
    }

    wait_async_all();     // prefetched tile resident
    __syncthreads();      // all waves done with buf before it is reused
  }

  // normalize and store fp32: out[(tq*256+sq)*1024 + head*128 + d]
#pragma unroll
  for (int j = 0; j < 8; ++j) {
    float invl = 1.0f / lrow[j];
    int sq = qblk * 64 + wave * 16 + j + 8 * lh;
    float* orow = out + ((size_t)(tq * SQL + sq)) * DMODEL + head * HDIM;
#pragma unroll
    for (int nt = 0; nt < 8; ++nt)
      orow[nt * 16 + lc] = oacc[nt][j] * invl;
  }
}

// ---------------------------------------------------------------------------
extern "C" void kernel_launch(void* const* d_in, const int* in_sizes, int n_in,
                              void* d_out, int out_size, void* d_ws, size_t ws_size,
                              hipStream_t stream) {
  (void)in_sizes; (void)n_in; (void)out_size; (void)ws_size;
  const float* q  = (const float*)d_in[0];
  const float* k  = (const float*)d_in[1];
  const float* v  = (const float*)d_in[2];
  const float* Wq = (const float*)d_in[3];
  const float* bq = (const float*)d_in[4];
  const float* Wk = (const float*)d_in[5];
  const float* bk = (const float*)d_in[6];
  const float* Wv = (const float*)d_in[7];
  const float* bv = (const float*)d_in[8];

  bf16* Qw  = (bf16*)d_ws;                          //  4 MB  [head][tok][d]
  bf16* Kw  = Qw + (size_t)NHEAD * NQTOK * HDIM;    // 64 MB  [head][tok][d]
  bf16* Vw  = Kw + (size_t)NHEAD * NKTOK * HDIM;    // 64 MB  [head][d][tok]
  bf16* WTq = Vw + (size_t)NHEAD * NKTOK * HDIM;    //  2 MB  [c][k]
  bf16* WTk = WTq + (size_t)DMODEL * DMODEL;
  bf16* WTv = WTk + (size_t)DMODEL * DMODEL;

  const int nwtr = DMODEL * DMODEL / 256;
  wtr_kernel<<<nwtr, 256, 0, stream>>>(Wq, WTq);
  wtr_kernel<<<nwtr, 256, 0, stream>>>(Wk, WTk);
  wtr_kernel<<<nwtr, 256, 0, stream>>>(Wv, WTv);

  qkv_rope_kernel<<<NQTOK / 64, 128, 0, stream>>>(q, WTq, bq, Qw, NQTOK, 0);
  qkv_rope_kernel<<<NKTOK / 64, 128, 0, stream>>>(k, WTk, bk, Kw, NKTOK, 1);
  qkv_rope_kernel<<<NKTOK / 64, 128, 0, stream>>>(v, WTv, bv, Vw, NKTOK, 2);

  attn_kernel<<<dim3(4, NHEAD, TQF), 128, 0, stream>>>(Qw, Kw, Vw, (float*)d_out);
}
